// ParallelMHA_46059229283014
// MI455X (gfx1250) — matrix operand order repaired and checked by reading, not compiled, not run on hardware
//
#include <hip/hip_runtime.h>

typedef __attribute__((ext_vector_type(16))) __bf16 v16bf;
typedef __attribute__((ext_vector_type(8)))  __bf16 v8bf;
typedef __attribute__((ext_vector_type(4)))  __bf16 v4bf;
typedef __attribute__((ext_vector_type(8)))  float  v8f;
typedef __attribute__((ext_vector_type(4)))  float  v4f;

union FragBF { v16bf v; v8bf h[2]; };

static __device__ inline v8f vzero8() {
    v8f v;
    #pragma unroll
    for (int i = 0; i < 8; ++i) v[i] = 0.0f;
    return v;
}

// CDNA5 async global->LDS copies (ASYNCcnt tracked). %0 = per-lane LDS byte
// address (low 32 bits of generic shared pointer), %1 = 64-bit global address.
#define ASYNC_CP32(ldsoff, gptr)                                              \
    asm volatile("global_load_async_to_lds_b128 %0, %1, off\n\t"              \
                 "global_load_async_to_lds_b128 %0, %1, off offset:16"        \
                 :: "v"(ldsoff), "v"(gptr) : "memory")

#define ASYNC_CP64(ldsoff, gptr)                                              \
    asm volatile("global_load_async_to_lds_b128 %0, %1, off\n\t"              \
                 "global_load_async_to_lds_b128 %0, %1, off offset:16\n\t"    \
                 "global_load_async_to_lds_b128 %0, %1, off offset:32\n\t"    \
                 "global_load_async_to_lds_b128 %0, %1, off offset:48"        \
                 :: "v"(ldsoff), "v"(gptr) : "memory")

#define ASYNC_WAIT() asm volatile("s_wait_asynccnt 0x0" ::: "memory")

// ---------------------------------------------------------------------------
// Elementwise fp32 -> bf16 (n4 = count/4)
// ---------------------------------------------------------------------------
__global__ __launch_bounds__(256) void f32_to_bf16(
    const float* __restrict__ in, __bf16* __restrict__ out, long n4)
{
    const long i = (long)blockIdx.x * 256 + threadIdx.x;
    if (i < n4) {
        const v4f x = *(const v4f*)(in + i * 4);
        v4bf y;
        #pragma unroll
        for (int r = 0; r < 4; ++r) y[r] = (__bf16)x[r];
        *(v4bf*)(out + i * 4) = y;
    }
}

// ---------------------------------------------------------------------------
// GEMM: out[M,N] = A[M,K] * W[N,K]^T + bias[N]  (bf16 in, f32 out, bf16 WMMA)
// Block: 256 threads (8 waves), 128x128 tile. Wave: 32x64 (2x4 WMMA accums).
// Double-buffered async global->LDS staging: tile k+1 copies overlap the
// 64 WMMAs of tile k.
// ---------------------------------------------------------------------------
__global__ __launch_bounds__(256) void gemm_xwT_bias(
    const __bf16* __restrict__ A, const __bf16* __restrict__ W,
    const float* __restrict__ bias, float* __restrict__ out,
    int M, int N, int K)
{
    __shared__ __align__(64) __bf16 lA[2][128][32];   // 16KB: A tiles, [row][k]
    __shared__ __align__(64) __bf16 lB[2][128][32];   // 16KB: W tiles, [n][k]
    const unsigned BUFSTRIDE = 128 * 32 * 2;          // bytes per buffer

    const int tid  = threadIdx.x;
    const int lane = tid & 31;
    const int wid  = tid >> 5;
    const int l    = lane & 15;
    const int hi   = lane >> 4;

    const int m0 = blockIdx.y * 128;
    const int n0 = blockIdx.x * 128;
    const int wm = (wid & 3) * 32;   // wave row offset in block tile
    const int wn = (wid >> 2) * 64;  // wave col offset in block tile

    v8f acc[2][4];
    #pragma unroll
    for (int i = 0; i < 2; ++i)
        #pragma unroll
        for (int j = 0; j < 4; ++j) acc[i][j] = vzero8();

    // staging: 256 threads, each copies 32B (16 bf16) per matrix per k-step
    const int sr = tid >> 1;          // 0..127
    const int sc = (tid & 1) * 16;    // 0 or 16
    const unsigned ldsA0 = (unsigned)(size_t)&lA[0][sr][sc];
    const unsigned ldsB0 = (unsigned)(size_t)&lB[0][sr][sc];
    const __bf16* gA = A + (size_t)(m0 + sr) * K + sc;
    const __bf16* gW = W + (size_t)(n0 + sr) * K + sc;

    // prologue: stage tile 0 into buffer 0
    ASYNC_CP32(ldsA0, (unsigned long long)(size_t)gA);
    ASYNC_CP32(ldsB0, (unsigned long long)(size_t)gW);

    int cur = 0;
    for (int k0 = 0; k0 < K; k0 += 32) {
        ASYNC_WAIT();          // my copies for buf[cur] have landed
        __syncthreads();       // everyone's landed; prior reads of buf[nxt] drained
        const int nxt = cur ^ 1;
        if (k0 + 32 < K) {     // overlap next tile's copies with this tile's WMMAs
            ASYNC_CP32(ldsA0 + (unsigned)nxt * BUFSTRIDE,
                       (unsigned long long)(size_t)(gA + k0 + 32));
            ASYNC_CP32(ldsB0 + (unsigned)nxt * BUFSTRIDE,
                       (unsigned long long)(size_t)(gW + k0 + 32));
        }

        // A fragments: lane holds row (l), K = hi*16 + s (16 contiguous, same order as B)
        FragBF af[2];
        #pragma unroll
        for (int i = 0; i < 2; ++i) {
            const __bf16* base = &lA[cur][wm + i * 16 + l][0];
            af[i].h[0] = *(const v8bf*)(base + hi * 16);       // K = hi*16 .. +8
            af[i].h[1] = *(const v8bf*)(base + hi * 16 + 8);   // K = hi*16+8 .. +8
        }
        // B fragments: lane holds column (l), K = hi*16 + j (16 contiguous)
        FragBF bf[4];
        #pragma unroll
        for (int j = 0; j < 4; ++j)
            bf[j].v = *(const v16bf*)(&lB[cur][wn + j * 16 + l][hi * 16]);

        #pragma unroll
        for (int i = 0; i < 2; ++i)
            #pragma unroll
            for (int j = 0; j < 4; ++j)
                acc[i][j] = __builtin_amdgcn_wmma_f32_16x16x32_bf16(
                    false, af[i].v, false, bf[j].v, (short)0, acc[i][j], false, false);
        cur = nxt;
    }

    // epilogue: C layout -> VGPR r holds (row = r + 8*hi, col = l)
    #pragma unroll
    for (int j = 0; j < 4; ++j) {
        const int col = n0 + wn + j * 16 + l;
        const float bv = bias[col];
        #pragma unroll
        for (int i = 0; i < 2; ++i) {
            #pragma unroll
            for (int r = 0; r < 8; ++r) {
                const int row = m0 + wm + i * 16 + r + 8 * hi;
                out[(size_t)row * N + col] = acc[i][j][r] + bv;
            }
        }
    }
}

// ---------------------------------------------------------------------------
// Rotary embedding on q,k + bf16 packing.
// q,k -> [B,H,S,D]; v -> [B,H,D,S] (transposed, so attention V staging is a
// straight async copy). 1/sqrt(D) folded into q. One block per (b,s,h).
// ---------------------------------------------------------------------------
__global__ __launch_bounds__(64) void rotary_pack(
    const float* __restrict__ qkv,   // [B,S,3,H,D]
    __bf16* __restrict__ qb, __bf16* __restrict__ kb, __bf16* __restrict__ vb)
{
    const int t = threadIdx.x;               // pair index 0..63
    int idx = blockIdx.x;                    // ((b*S)+s)*H + h
    const int h = idx & 15; idx >>= 4;
    const int s = idx & 2047;
    const int b = idx >> 11;
    const int bh = b * 16 + h;

    const size_t base = (((size_t)(b * 2048 + s) * 3) * 16 + h) * 128;
    const float q1 = qkv[base + t],        q2 = qkv[base + 64 + t];
    const float k1 = qkv[base + 2048 + t], k2 = qkv[base + 2048 + 64 + t];
    const float v1 = qkv[base + 4096 + t], v2 = qkv[base + 4096 + 64 + t];

    // inv_freq = 10000^(-t/64); freqs = s * inv_freq
    const float inv = __expf(-(float)t * (9.210340371976184f / 64.0f));
    float sn, cs;
    __sincosf((float)s * inv, &sn, &cs);
    const float scale = 0.08838834764831845f; // 1/sqrt(128)

    const size_t o = ((size_t)bh * 2048 + s) * 128 + t;     // [B,H,S,D]
    qb[o]      = (__bf16)((q1 * cs - q2 * sn) * scale);
    qb[o + 64] = (__bf16)((q2 * cs + q1 * sn) * scale);
    kb[o]      = (__bf16)(k1 * cs - k2 * sn);
    kb[o + 64] = (__bf16)(k2 * cs + k1 * sn);

    const size_t ov = ((size_t)bh * 128 + t) * 2048 + s;    // [B,H,D,S]
    vb[ov]              = (__bf16)v1;
    vb[ov + 64 * 2048]  = (__bf16)v2;
}

// ---------------------------------------------------------------------------
// Causal flash attention. Grid: (S/64, B*H). Block: 128 threads (4 waves).
// Each wave owns 16 query rows; 32-key chunks, online softmax.
// Double-buffered async K/V staging overlaps the next chunk's copies with the
// current chunk's 16 WMMAs + softmax.
// ---------------------------------------------------------------------------
__global__ __launch_bounds__(128) void flash_attn(
    const __bf16* __restrict__ qb, const __bf16* __restrict__ kb,
    const __bf16* __restrict__ vbT, __bf16* __restrict__ ctx)
{
    __shared__ __align__(64) __bf16 sK[2][32][128];   // 16KB  [key][d]
    __shared__ __align__(64) __bf16 sVt[2][128][32];  // 16KB  [d][key]
    __shared__ float  sS[4][16][32];                  // 8KB   per-wave score tile
    __shared__ __align__(64) __bf16 sP[4][16][32];    // 4KB   per-wave P (A-frag)
    __shared__ float  sAlpha[4][16];
    __shared__ float  sLsum[4][16];
    const unsigned BUFSTRIDE = 32 * 128 * 2;          // bytes per K/V buffer

    const int tid  = threadIdx.x;
    const int lane = tid & 31;
    const int wid  = tid >> 5;
    const int l    = lane & 15;
    const int hi   = lane >> 4;

    const int bh  = blockIdx.y;            // b*16 + h
    const int q0b = blockIdx.x * 64;
    const int q0w = q0b + wid * 16;

    const __bf16* Q  = qb  + (size_t)bh * 2048 * 128;
    const __bf16* Kp = kb  + (size_t)bh * 2048 * 128;
    const __bf16* Vt = vbT + (size_t)bh * 128 * 2048;

    // Q fragments for D=128 (4 x 16x32 A-frags), kept in registers
    FragBF qf[4];
    #pragma unroll
    for (int c = 0; c < 4; ++c) {
        const __bf16* base = Q + (size_t)(q0w + l) * 128 + c * 32;
        qf[c].h[0] = *(const v8bf*)(base + hi * 16);
        qf[c].h[1] = *(const v8bf*)(base + hi * 16 + 8);
    }

    v8f cacc[8];
    #pragma unroll
    for (int t = 0; t < 8; ++t) cacc[t] = vzero8();

    float mrun = -1e30f, lrun = 0.0f;      // per-row state (lanes 0..15, row = l)

    // async staging: K: thread -> (key row kr, 64B of d); V: thread -> d row tid
    const int kr = tid >> 2;
    const int dc = (tid & 3) * 32;
    const unsigned ldsK0 = (unsigned)(size_t)&sK[0][kr][dc];
    const unsigned ldsV0 = (unsigned)(size_t)&sVt[0][tid][0];
    const __bf16* gK = Kp + (size_t)kr * 128 + dc;   // + kc*128
    const __bf16* gV = Vt + (size_t)tid * 2048;      // + kc

    const int kend = q0b + 64;             // causal: keys < q0b+64
    // prologue: stage chunk 0 into buffer 0
    ASYNC_CP64(ldsK0, (unsigned long long)(size_t)gK);
    ASYNC_CP64(ldsV0, (unsigned long long)(size_t)gV);

    int cur = 0;
    for (int kc = 0; kc < kend; kc += 32) {
        ASYNC_WAIT();
        __syncthreads();
        const int nxt = cur ^ 1;
        if (kc + 32 < kend) {   // overlap next chunk's copies with this chunk's work
            ASYNC_CP64(ldsK0 + (unsigned)nxt * BUFSTRIDE,
                       (unsigned long long)(size_t)(gK + (size_t)(kc + 32) * 128));
            ASYNC_CP64(ldsV0 + (unsigned)nxt * BUFSTRIDE,
                       (unsigned long long)(size_t)(gV + kc + 32));
        }

        // S = Q * K^T : 2 key n-tiles x 4 d-chunks of WMMA
        v8f sacc[2];
        sacc[0] = vzero8(); sacc[1] = vzero8();
        #pragma unroll
        for (int nt = 0; nt < 2; ++nt) {
            #pragma unroll
            for (int c = 0; c < 4; ++c) {
                v16bf kf = *(const v16bf*)(&sK[cur][nt * 16 + l][c * 32 + hi * 16]);
                sacc[nt] = __builtin_amdgcn_wmma_f32_16x16x32_bf16(
                    false, qf[c].v, false, kf, (short)0, sacc[nt], false, false);
            }
        }
        #pragma unroll
        for (int nt = 0; nt < 2; ++nt)
            #pragma unroll
            for (int r = 0; r < 8; ++r)
                sS[wid][r + 8 * hi][nt * 16 + l] = sacc[nt][r];
        __syncthreads();

        // online softmax: lane r<16 owns row r
        if (lane < 16) {
            const int qi = q0w + l;
            float mx = mrun;
            float sc[32];
            #pragma unroll
            for (int j = 0; j < 32; ++j) {
                float s = sS[wid][l][j];
                if (kc + j > qi) s = -1e30f;
                sc[j] = s;
                mx = fmaxf(mx, s);
            }
            const float alpha = __expf(mrun - mx);
            float ps = 0.0f;
            #pragma unroll
            for (int j = 0; j < 32; ++j) {
                const float p = __expf(sc[j] - mx);
                ps += p;
                sP[wid][l][j] = (__bf16)p;
            }
            lrun = lrun * alpha + ps;
            mrun = mx;
            sAlpha[wid][l] = alpha;
        }
        __syncthreads();

        // rescale accumulators by per-row alpha (row = r + 8*hi)
        v8f av;
        {
            const float* ab = &sAlpha[wid][hi * 8];
            const v4f a0 = *(const v4f*)(ab);
            const v4f a1 = *(const v4f*)(ab + 4);
            #pragma unroll
            for (int r = 0; r < 4; ++r) { av[r] = a0[r]; av[r + 4] = a1[r]; }
        }
        #pragma unroll
        for (int t = 0; t < 8; ++t)
            #pragma unroll
            for (int r = 0; r < 8; ++r)
                cacc[t][r] *= av[r];

        // ctx += P * V : 8 d n-tiles of WMMA
        FragBF pf;
        {
            const __bf16* base = &sP[wid][l][0];
            pf.h[0] = *(const v8bf*)(base + hi * 16);
            pf.h[1] = *(const v8bf*)(base + hi * 16 + 8);
        }
        #pragma unroll
        for (int t = 0; t < 8; ++t) {
            v16bf vf = *(const v16bf*)(&sVt[cur][t * 16 + l][hi * 16]);
            cacc[t] = __builtin_amdgcn_wmma_f32_16x16x32_bf16(
                false, pf.v, false, vf, (short)0, cacc[t], false, false);
        }
        cur = nxt;
    }

    __syncthreads();
    if (lane < 16) sLsum[wid][l] = lrun;
    __syncthreads();
    v8f lv;
    {
        const float* lb = &sLsum[wid][hi * 8];
        const v4f a0 = *(const v4f*)(lb);
        const v4f a1 = *(const v4f*)(lb + 4);
        #pragma unroll
        for (int r = 0; r < 4; ++r) { lv[r] = 1.0f / a0[r]; lv[r + 4] = 1.0f / a1[r]; }
    }

    // write ctx bf16 [B,S,E], E-col = h*128 + d
    const int b = bh >> 4, h = bh & 15;
    #pragma unroll
    for (int t = 0; t < 8; ++t) {
        #pragma unroll
        for (int r = 0; r < 8; ++r) {
            const int srow = q0w + r + 8 * hi;
            const int col  = h * 128 + t * 16 + l;
            ctx[(size_t)(b * 2048 + srow) * 2048 + col] = (__bf16)(cacc[t][r] * lv[r]);
        }
    }
}

// ---------------------------------------------------------------------------
extern "C" void kernel_launch(void* const* d_in, const int* in_sizes, int n_in,
                              void* d_out, int out_size, void* d_ws, size_t ws_size,
                              hipStream_t stream) {
    const float* x    = (const float*)d_in[0];
    const float* wqkv = (const float*)d_in[1];
    const float* bqkv = (const float*)d_in[2];
    const float* wout = (const float*)d_in[3];
    const float* bout = (const float*)d_in[4];
    float* out = (float*)d_out;

    const int B = 2, S = 2048, E = 2048, H = 16, D = 128;
    const int M = B * S;            // 4096
    const int F = 3 * E;            // 6144

    char* ws = (char*)d_ws;
    size_t off = 0;
    float*  qkv   = (float*)(ws + off);  off += (size_t)M * F * sizeof(float);
    __bf16* qb    = (__bf16*)(ws + off); off += (size_t)B * H * S * D * sizeof(__bf16);
    __bf16* kb    = (__bf16*)(ws + off); off += (size_t)B * H * S * D * sizeof(__bf16);
    __bf16* vbT   = (__bf16*)(ws + off); off += (size_t)B * H * S * D * sizeof(__bf16);
    __bf16* xb    = (__bf16*)(ws + off); off += (size_t)M * E * sizeof(__bf16);
    __bf16* wqkvb = (__bf16*)(ws + off); off += (size_t)F * E * sizeof(__bf16);
    __bf16* woutb = (__bf16*)(ws + off); off += (size_t)E * E * sizeof(__bf16);
    __bf16* ctxb  = (__bf16*)(ws + off); off += (size_t)M * E * sizeof(__bf16);

    // 0) bf16 conversions of GEMM operands
    {
        long n4x = (long)M * E / 4, n4w = (long)F * E / 4, n4o = (long)E * E / 4;
        f32_to_bf16<<<dim3((unsigned)((n4x + 255) / 256)), 256, 0, stream>>>(x, xb, n4x);
        f32_to_bf16<<<dim3((unsigned)((n4w + 255) / 256)), 256, 0, stream>>>(wqkv, wqkvb, n4w);
        f32_to_bf16<<<dim3((unsigned)((n4o + 255) / 256)), 256, 0, stream>>>(wout, woutb, n4o);
    }
    // 1) qkv = x @ wqkv^T + bqkv
    gemm_xwT_bias<<<dim3(F / 128, M / 128), 256, 0, stream>>>(xb, wqkvb, bqkv, qkv, M, F, E);
    // 2) rotary + bf16 pack (V transposed)
    rotary_pack<<<dim3(B * S * H), 64, 0, stream>>>(qkv, qb, kb, vbT);
    // 3) causal flash attention -> bf16 ctx
    flash_attn<<<dim3(S / 64, B * H), 128, 0, stream>>>(qb, kb, vbT, ctxb);
    // 4) out = ctx @ wout^T + bout
    gemm_xwT_bias<<<dim3(E / 128, M / 128), 256, 0, stream>>>(ctxb, woutb, bout, out, M, E, E);
}
